// Swin_50886772523389
// MI455X (gfx1250) — matrix operand order repaired and checked by reading, not compiled, not run on hardware
//
#include <hip/hip_runtime.h>
#include <hip/hip_bf16.h>

typedef _Float16 half_t;
typedef __attribute__((ext_vector_type(8)))  _Float16 v8h;
typedef __attribute__((ext_vector_type(16))) _Float16 v16h;
typedef __attribute__((ext_vector_type(8)))  float    v8f;

#define WMMA_F16(a, b, c) \
  __builtin_amdgcn_wmma_f32_16x16x32_f16(false, (a), false, (b), (short)0, (c), false, false)

__device__ __forceinline__ v16h cat16(v8h lo, v8h hi) {
  v16h r;
#pragma unroll
  for (int i = 0; i < 8; ++i) { r[i] = lo[i]; r[i + 8] = hi[i]; }
  return r;
}

// A operand tile: 16(M) x 32(K), src row-major, ld in halves, src points at tile origin.
__device__ __forceinline__ v16h load_a16x32(const half_t* __restrict__ src, int ld, int lane) {
  const int m  = lane & 15;
  const int ks = (lane >> 4) << 3;            // 0 or 8
  const half_t* p = src + m * ld + ks;
  v8h lo = *(const v8h*)(p);                  // K = ks + 0..7
  v8h hi = *(const v8h*)(p + 16);             // K = 16 + ks + 0..7
  return cat16(lo, hi);
}

// B operand tile: 32(K) x 16(N), where B(k,n) = W[n*ld + k]  (W stored [N][K] row-major).
__device__ __forceinline__ v16h load_b32x16(const half_t* __restrict__ wm, int ld, int lane) {
  const int n  = lane & 15;
  const int kb = (lane >> 4) << 3;            // 0 or 8
  const half_t* p = wm + n * ld + kb;
  v8h lo = *(const v8h*)(p);                  // K = kb + 0..7
  v8h hi = *(const v8h*)(p + 16);             // K = 16 + kb + 0..7
  return cat16(lo, hi);
}
// C/D layout (f32 16x16): lane l, reg r -> M = r + 8*(l>>4), N = l&15.

// ---- CDNA5 async global->LDS copy (ASYNCcnt path), 16 B per lane per issue ----
__device__ __forceinline__ void async_cp16(unsigned lds_off, const void* gptr) {
  asm volatile("global_load_async_to_lds_b128 %0, %1, off"
               :: "v"(lds_off), "v"(gptr) : "memory");
}
__device__ __forceinline__ void async_wait0() {
  asm volatile("s_wait_asynccnt 0" ::: "memory");
}

// Stage 64x256 halves (32 KB) contiguous from src into sA via async-to-LDS.
__device__ __forceinline__ void stage32k(half_t* sA, const half_t* src, int tid) {
  const unsigned l0 = (unsigned)(uintptr_t)(void*)sA;
  for (int i = tid; i < 2048; i += 256)
    async_cp16(l0 + (unsigned)i * 16u, (const void*)((const char*)src + (size_t)i * 16));
  async_wait0();
}

__global__ __launch_bounds__(256) void k_cvt(const float* __restrict__ s,
                                             half_t* __restrict__ d, int n) {
  int i = blockIdx.x * 256 + threadIdx.x;
  if (i < n) d[i] = (half_t)s[i];
}

// ---- LN1 + NCHW->token + window partition; one block per token ----
__global__ __launch_bounds__(256) void k_ln1_win(const float* __restrict__ x,
                                                 const float* __restrict__ g,
                                                 const float* __restrict__ bt,
                                                 float* __restrict__ tok,
                                                 half_t* __restrict__ tn) {
  const int t = blockIdx.x;                   // 0..65535
  const int c = threadIdx.x;                  // 0..255
  const int b = t >> 12, hw = t & 4095;
  const int h = hw >> 6, w = hw & 63;
  float val = x[(size_t)(b * 256 + c) * 4096 + hw];
  tok[(size_t)t * 256 + c] = val;
  __shared__ float red[256];
  red[c] = val; __syncthreads();
  for (int s = 128; s > 0; s >>= 1) { if (c < s) red[c] += red[c + s]; __syncthreads(); }
  float mean = red[0] * (1.f / 256.f);
  __syncthreads();
  float d = val - mean;
  red[c] = d * d; __syncthreads();
  for (int s = 128; s > 0; s >>= 1) { if (c < s) red[c] += red[c + s]; __syncthreads(); }
  float var = red[0] * (1.f / 256.f);
  float nv = d * rsqrtf(var + 1e-5f) * g[c] + bt[c];
  const int win = (b << 6) | ((h >> 3) << 3) | (w >> 3);
  const int nid = ((h & 7) << 3) | (w & 7);
  tn[((size_t)win * 64 + nid) * 256 + c] = (half_t)nv;
}

// ---- QKV GEMM: per window M=64, N=768, K=256; wave = 4Mx2N macro-tile ----
__global__ __launch_bounds__(256) void k_qkv(const half_t* __restrict__ tn,
                                             const half_t* __restrict__ wq,
                                             const float* __restrict__ bq,
                                             half_t* __restrict__ qo,
                                             half_t* __restrict__ ko,
                                             half_t* __restrict__ vo) {
  __shared__ half_t sA[64 * 256];
  const int tid = threadIdx.x, win = blockIdx.x;
  stage32k(sA, tn + (size_t)win * 64 * 256, tid);
  __syncthreads();
  const int wave = tid >> 5, lane = tid & 31;
  const int nc = lane & 15, lg = lane >> 4;
  for (int pp = wave; pp < 24; pp += 8) {     // 24 N-pairs (48 N-tiles)
    const int nt0 = pp * 2;
    v8f acc[4][2];
#pragma unroll
    for (int mt = 0; mt < 4; ++mt) { acc[mt][0] = {}; acc[mt][1] = {}; }
#pragma unroll
    for (int kk = 0; kk < 256; kk += 32) {
      v16h b0 = load_b32x16(wq + (size_t)nt0 * 16 * 256 + kk, 256, lane);
      v16h b1 = load_b32x16(wq + (size_t)(nt0 + 1) * 16 * 256 + kk, 256, lane);
#pragma unroll
      for (int mt = 0; mt < 4; ++mt) {
        v16h av = load_a16x32(sA + mt * 16 * 256 + kk, 256, lane);
        acc[mt][0] = WMMA_F16(av, b0, acc[mt][0]);
        acc[mt][1] = WMMA_F16(av, b1, acc[mt][1]);
      }
    }
#pragma unroll
    for (int j = 0; j < 2; ++j) {
      const int n = (nt0 + j) * 16 + nc;
      const float bias = bq[n];
      const int which = n >> 8, head = (n >> 5) & 7, d = n & 31;
      const float scale = (which == 0) ? 0.1767766953f : 1.f;   // 1/sqrt(32) on q
#pragma unroll
      for (int mt = 0; mt < 4; ++mt)
#pragma unroll
        for (int r = 0; r < 8; ++r) {
          const int m = mt * 16 + r + (lg << 3);
          float v = (acc[mt][j][r] + bias) * scale;
          if (which == 0)      qo[(((size_t)win * 8 + head) * 64 + m) * 32 + d] = (half_t)v;
          else if (which == 1) ko[(((size_t)win * 8 + head) * 64 + m) * 32 + d] = (half_t)v;
          else                 vo[(((size_t)win * 8 + head) * 32 + d) * 64 + m] = (half_t)v;
        }
    }
  }
}

// ---- Attention: one wave per (window, head) ----
__global__ __launch_bounds__(128) void k_attn(const half_t* __restrict__ q,
                                              const half_t* __restrict__ kbuf,
                                              const half_t* __restrict__ vT,
                                              const float* __restrict__ rpb,
                                              const int* __restrict__ rpi,
                                              half_t* __restrict__ ao) {
  __shared__ half_t sP[4][64 * 64];           // 32 KB: per-wave P scratch
  const int tid = threadIdx.x, wave = tid >> 5, lane = tid & 31;
  const int win = blockIdx.x >> 1;
  const int head = ((blockIdx.x & 1) << 2) + wave;
  const half_t* qh = q    + ((size_t)win * 8 + head) * 64 * 32;
  const half_t* kh = kbuf + ((size_t)win * 8 + head) * 64 * 32;
  const half_t* vh = vT   + ((size_t)win * 8 + head) * 32 * 64;
  const int nc = lane & 15, lg = lane >> 4;

  // S = q k^T : preload 4 K-operand tiles, reuse across mi
  v16h bk[4];
#pragma unroll
  for (int ni = 0; ni < 4; ++ni) bk[ni] = load_b32x16(kh + ni * 16 * 32, 32, lane);
  v8f s[4][4];
#pragma unroll
  for (int mi = 0; mi < 4; ++mi) {
    v16h av = load_a16x32(qh + mi * 16 * 32, 32, lane);
#pragma unroll
    for (int ni = 0; ni < 4; ++ni) {
      v8f z = {};
      s[mi][ni] = WMMA_F16(av, bk[ni], z);
    }
  }
  // relative position bias
#pragma unroll
  for (int mi = 0; mi < 4; ++mi)
#pragma unroll
    for (int ni = 0; ni < 4; ++ni)
#pragma unroll
      for (int r = 0; r < 8; ++r) {
        int row = mi * 16 + r + (lg << 3);
        int col = ni * 16 + nc;
        s[mi][ni][r] += rpb[rpi[row * 64 + col] * 8 + head];
      }
  // softmax over 64 cols: cols live on 16 lanes x 4 ni-tiles -> half-wave xor reduce
#pragma unroll
  for (int mi = 0; mi < 4; ++mi) {
    float inv[8];
#pragma unroll
    for (int r = 0; r < 8; ++r) {
      float mx = fmaxf(fmaxf(s[mi][0][r], s[mi][1][r]), fmaxf(s[mi][2][r], s[mi][3][r]));
#pragma unroll
      for (int off = 1; off < 16; off <<= 1) mx = fmaxf(mx, __shfl_xor(mx, off, 32));
      float sum = 0.f;
#pragma unroll
      for (int ni = 0; ni < 4; ++ni) {
        float e = __expf(s[mi][ni][r] - mx);
        s[mi][ni][r] = e; sum += e;
      }
#pragma unroll
      for (int off = 1; off < 16; off <<= 1) sum += __shfl_xor(sum, off, 32);
      inv[r] = 1.f / sum;
    }
#pragma unroll
    for (int ni = 0; ni < 4; ++ni)
#pragma unroll
      for (int r = 0; r < 8; ++r)
        sP[wave][(mi * 16 + r + (lg << 3)) * 64 + ni * 16 + nc] =
            (half_t)(s[mi][ni][r] * inv[r]);
  }
  // O = P @ V: preload the 4 V-operand tiles [nd][kt], reuse across mi
  v16h bvv[2][2];
#pragma unroll
  for (int nd = 0; nd < 2; ++nd)
#pragma unroll
    for (int kt = 0; kt < 2; ++kt)
      bvv[nd][kt] = load_b32x16(vh + nd * 16 * 64 + kt * 32, 64, lane);
#pragma unroll
  for (int mi = 0; mi < 4; ++mi) {
    v8f acc[2]; acc[0] = {}; acc[1] = {};
#pragma unroll
    for (int kt = 0; kt < 2; ++kt) {
      v16h av = load_a16x32(&sP[wave][mi * 16 * 64 + kt * 32], 64, lane);
      acc[0] = WMMA_F16(av, bvv[0][kt], acc[0]);
      acc[1] = WMMA_F16(av, bvv[1][kt], acc[1]);
    }
#pragma unroll
    for (int nd = 0; nd < 2; ++nd) {
      const int c = head * 32 + nd * 16 + nc;
#pragma unroll
      for (int r = 0; r < 8; ++r) {
        int n = mi * 16 + r + (lg << 3);
        ao[((size_t)win * 64 + n) * 256 + c] = (half_t)acc[nd][r];
      }
    }
  }
}

// ---- Proj GEMM + residual: per window M=64, N=256, K=256; wave = 4Mx2N ----
__global__ __launch_bounds__(256) void k_proj(const half_t* __restrict__ ao,
                                              const half_t* __restrict__ wp,
                                              const float* __restrict__ bp,
                                              float* __restrict__ tok) {
  __shared__ half_t sA[64 * 256];
  const int tid = threadIdx.x, win = blockIdx.x;
  stage32k(sA, ao + (size_t)win * 64 * 256, tid);
  __syncthreads();
  const int wave = tid >> 5, lane = tid & 31;
  const int nc = lane & 15, lg = lane >> 4;
  const int b = win >> 6, wy = (win >> 3) & 7, wx = win & 7;
  const int nt0 = wave * 2;                    // 8 N-pairs, one per wave
  v8f acc[4][2];
#pragma unroll
  for (int mt = 0; mt < 4; ++mt) { acc[mt][0] = {}; acc[mt][1] = {}; }
#pragma unroll
  for (int kk = 0; kk < 256; kk += 32) {
    v16h b0 = load_b32x16(wp + (size_t)nt0 * 16 * 256 + kk, 256, lane);
    v16h b1 = load_b32x16(wp + (size_t)(nt0 + 1) * 16 * 256 + kk, 256, lane);
#pragma unroll
    for (int mt = 0; mt < 4; ++mt) {
      v16h av = load_a16x32(sA + mt * 16 * 256 + kk, 256, lane);
      acc[mt][0] = WMMA_F16(av, b0, acc[mt][0]);
      acc[mt][1] = WMMA_F16(av, b1, acc[mt][1]);
    }
  }
#pragma unroll
  for (int j = 0; j < 2; ++j) {
    const int c = (nt0 + j) * 16 + nc;
    const float bias = bp[c];
#pragma unroll
    for (int mt = 0; mt < 4; ++mt)
#pragma unroll
      for (int r = 0; r < 8; ++r) {
        const int m = mt * 16 + r + (lg << 3);
        const int h = wy * 8 + (m >> 3), w = wx * 8 + (m & 7);
        const int gt = (b << 12) + (h << 6) + w;
        tok[(size_t)gt * 256 + c] += acc[mt][j][r] + bias;
      }
  }
}

// ---- LN2: one block per token ----
__global__ __launch_bounds__(256) void k_ln2k(const float* __restrict__ tok,
                                              const float* __restrict__ g,
                                              const float* __restrict__ bt,
                                              half_t* __restrict__ t2) {
  const int t = blockIdx.x, c = threadIdx.x;
  float val = tok[(size_t)t * 256 + c];
  __shared__ float red[256];
  red[c] = val; __syncthreads();
  for (int s = 128; s > 0; s >>= 1) { if (c < s) red[c] += red[c + s]; __syncthreads(); }
  float mean = red[0] * (1.f / 256.f);
  __syncthreads();
  float d = val - mean;
  red[c] = d * d; __syncthreads();
  for (int s = 128; s > 0; s >>= 1) { if (c < s) red[c] += red[c + s]; __syncthreads(); }
  float var = red[0] * (1.f / 256.f);
  t2[(size_t)t * 256 + c] = (half_t)(d * rsqrtf(var + 1e-5f) * g[c] + bt[c]);
}

// ---- FFN1 + exact GELU: per 64-token group M=64, N=1024, K=256; wave = 4Mx2N x4 ----
__global__ __launch_bounds__(256) void k_ffn1(const half_t* __restrict__ t2,
                                              const half_t* __restrict__ w1,
                                              const float* __restrict__ b1,
                                              half_t* __restrict__ hb) {
  __shared__ half_t sA[64 * 256];
  const int tid = threadIdx.x, grp = blockIdx.x;
  stage32k(sA, t2 + (size_t)grp * 64 * 256, tid);
  __syncthreads();
  const int wave = tid >> 5, lane = tid & 31;
  const int nc = lane & 15, lg = lane >> 4;
  for (int pp = wave; pp < 32; pp += 8) {     // 32 N-pairs (64 N-tiles)
    const int nt0 = pp * 2;
    v8f acc[4][2];
#pragma unroll
    for (int mt = 0; mt < 4; ++mt) { acc[mt][0] = {}; acc[mt][1] = {}; }
#pragma unroll
    for (int kk = 0; kk < 256; kk += 32) {
      v16h b0 = load_b32x16(w1 + (size_t)nt0 * 16 * 256 + kk, 256, lane);
      v16h b1 = load_b32x16(w1 + (size_t)(nt0 + 1) * 16 * 256 + kk, 256, lane);
#pragma unroll
      for (int mt = 0; mt < 4; ++mt) {
        v16h av = load_a16x32(sA + mt * 16 * 256 + kk, 256, lane);
        acc[mt][0] = WMMA_F16(av, b0, acc[mt][0]);
        acc[mt][1] = WMMA_F16(av, b1, acc[mt][1]);
      }
    }
#pragma unroll
    for (int j = 0; j < 2; ++j) {
      const int n = (nt0 + j) * 16 + nc;
      const float bias = b1[n];
#pragma unroll
      for (int mt = 0; mt < 4; ++mt)
#pragma unroll
        for (int r = 0; r < 8; ++r) {
          const int m = mt * 16 + r + (lg << 3);
          const int gt = grp * 64 + m;
          float xv = acc[mt][j][r] + bias;
          float gl = 0.5f * xv * (1.f + erff(xv * 0.70710678f));   // exact GELU
          hb[(size_t)gt * 1024 + n] = (half_t)gl;
        }
    }
  }
}

// ---- FFN2 + residual + token->NCHW: per group M=64, N=256, K=1024 (4 LDS chunks) ----
__global__ __launch_bounds__(256) void k_ffn2(const half_t* __restrict__ hb,
                                              const half_t* __restrict__ w2,
                                              const float* __restrict__ b2,
                                              const float* __restrict__ tok,
                                              float* __restrict__ out) {
  __shared__ half_t sA[64 * 256];
  const int tid = threadIdx.x, grp = blockIdx.x;
  const int wave = tid >> 5, lane = tid & 31;
  const int nc = lane & 15, lg = lane >> 4;
  const int nt0 = wave * 2;                    // one N-pair per wave
  v8f acc[4][2];
#pragma unroll
  for (int mt = 0; mt < 4; ++mt) { acc[mt][0] = {}; acc[mt][1] = {}; }
  const unsigned l0 = (unsigned)(uintptr_t)(void*)sA;
  for (int kc = 0; kc < 4; ++kc) {
    for (int i = tid; i < 2048; i += 256) {
      const int row = i >> 5, col = (i & 31) << 3;        // col in halves
      async_cp16(l0 + (unsigned)i * 16u,
                 (const void*)(hb + (size_t)(grp * 64 + row) * 1024 + kc * 256 + col));
    }
    async_wait0();
    __syncthreads();
#pragma unroll
    for (int kk = 0; kk < 256; kk += 32) {
      v16h b0 = load_b32x16(w2 + (size_t)nt0 * 16 * 1024 + kc * 256 + kk, 1024, lane);
      v16h b1 = load_b32x16(w2 + (size_t)(nt0 + 1) * 16 * 1024 + kc * 256 + kk, 1024, lane);
#pragma unroll
      for (int mt = 0; mt < 4; ++mt) {
        v16h av = load_a16x32(sA + mt * 16 * 256 + kk, 256, lane);
        acc[mt][0] = WMMA_F16(av, b0, acc[mt][0]);
        acc[mt][1] = WMMA_F16(av, b1, acc[mt][1]);
      }
    }
    __syncthreads();
  }
#pragma unroll
  for (int j = 0; j < 2; ++j) {
    const int c = (nt0 + j) * 16 + nc;
    const float bias = b2[c];
#pragma unroll
    for (int mt = 0; mt < 4; ++mt)
#pragma unroll
      for (int r = 0; r < 8; ++r) {
        const int m = mt * 16 + r + (lg << 3);
        const int gt = grp * 64 + m;
        float v = tok[(size_t)gt * 256 + c] + bias + acc[mt][j][r];
        const int b = gt >> 12, hw = gt & 4095;
        out[(size_t)(b * 256 + c) * 4096 + hw] = v;
      }
  }
}

extern "C" void kernel_launch(void* const* d_in, const int* in_sizes, int n_in,
                              void* d_out, int out_size, void* d_ws, size_t ws_size,
                              hipStream_t stream) {
  (void)in_sizes; (void)n_in; (void)out_size; (void)ws_size;
  const float* x      = (const float*)d_in[0];
  const float* ln1_g  = (const float*)d_in[1];
  const float* ln1_b  = (const float*)d_in[2];
  const float* qkv_w  = (const float*)d_in[3];
  const float* qkv_b  = (const float*)d_in[4];
  const float* proj_w = (const float*)d_in[5];
  const float* proj_b = (const float*)d_in[6];
  const float* rpb    = (const float*)d_in[7];
  const float* ln2_g  = (const float*)d_in[8];
  const float* ln2_b  = (const float*)d_in[9];
  const float* w1     = (const float*)d_in[10];
  const float* b1     = (const float*)d_in[11];
  const float* w2     = (const float*)d_in[12];
  const float* b2     = (const float*)d_in[13];
  const int*   rpi    = (const int*)d_in[14];
  float* out = (float*)d_out;

  char* p = (char*)d_ws;
  half_t* wqkv_h  = (half_t*)p; p += (size_t)768 * 256 * 2;
  half_t* wproj_h = (half_t*)p; p += (size_t)256 * 256 * 2;
  half_t* w1_h    = (half_t*)p; p += (size_t)1024 * 256 * 2;
  half_t* w2_h    = (half_t*)p; p += (size_t)256 * 1024 * 2;
  float*  tok     = (float*)p;  p += (size_t)65536 * 256 * 4;
  half_t* big     = (half_t*)p; p += (size_t)65536 * 1024 * 2;   // tnorm|q|k|vT -> reused as hb
  half_t* t2      = (half_t*)p; p += (size_t)65536 * 256 * 2;

  half_t* tn = big;
  half_t* qb = big + (size_t)65536 * 256;
  half_t* kb = big + (size_t)2 * 65536 * 256;
  half_t* vb = big + (size_t)3 * 65536 * 256;
  half_t* ao = tn;    // tnorm dead after k_qkv
  half_t* hb = big;   // whole region dead after k_proj

  k_cvt<<<768, 256, 0, stream>>>(qkv_w, wqkv_h, 768 * 256);
  k_cvt<<<256, 256, 0, stream>>>(proj_w, wproj_h, 256 * 256);
  k_cvt<<<1024, 256, 0, stream>>>(w1, w1_h, 1024 * 256);
  k_cvt<<<1024, 256, 0, stream>>>(w2, w2_h, 1024 * 256);
  k_ln1_win<<<65536, 256, 0, stream>>>(x, ln1_g, ln1_b, tok, tn);
  k_qkv<<<1024, 256, 0, stream>>>(tn, wqkv_h, qkv_b, qb, kb, vb);
  k_attn<<<2048, 128, 0, stream>>>(qb, kb, vb, rpb, rpi, ao);
  k_proj<<<1024, 256, 0, stream>>>(ao, wproj_h, proj_b, tok);
  k_ln2k<<<65536, 256, 0, stream>>>(tok, ln2_g, ln2_b, t2);
  k_ffn1<<<1024, 256, 0, stream>>>(t2, w1_h, b1, hb);
  k_ffn2<<<1024, 256, 0, stream>>>(hb, w2_h, b2, tok, out);
}
